// MolecularGraphNeuralNetwork_65369402245720
// MI455X (gfx1250) — matrix operand order repaired and checked by reading, not compiled, not run on hardware
//
#include <hip/hip_runtime.h>
#include <hip/hip_bf16.h>

// MI455X / gfx1250 fused molecular-GNN forward.
// One workgroup per molecule: TDM-loaded adjacency block + async embed gather
// into LDS, 3 GNN layers with v_wmma_f32_16x16x32_f16 (wave32, 8 waves/block),
// l2norm via wave32 shfl reductions, sum-pool, then WMMA output MLP + MSE loss.
// Block-diagonal adjacency read: 2x512KB instead of 2x64MB.

#define N_MOLV 128
#define ATOMSV 32
#define N_ATV  4096
#define DIMV   64
#define FEATV  187
#define FPADV  192
#define NLAYER 3

typedef __attribute__((ext_vector_type(16))) _Float16 v16h;
typedef __attribute__((ext_vector_type(8)))  _Float16 v8h;
typedef __attribute__((ext_vector_type(8)))  float    v8f;
typedef __attribute__((ext_vector_type(4)))  unsigned int v4u;
typedef __attribute__((ext_vector_type(8)))  int      v8i;
typedef __attribute__((ext_vector_type(4)))  int      v4i;

#if __has_builtin(__builtin_amdgcn_tensor_load_to_lds)
#define HAVE_TDM 1
#else
#define HAVE_TDM 0
#endif
#if __has_builtin(__builtin_amdgcn_global_load_async_to_lds_b32)
#define HAVE_ASYNC 1
#else
#define HAVE_ASYNC 0
#endif

// Build the 16x32 f16 A-tile fragment for this lane from a row-major row
// pointer: elements [0..15] = K[c0..c0+15].
__device__ __forceinline__ v16h load_a_frag(const _Float16* rowPtr, int c0) {
    v8h lo = *(const v8h*)(rowPtr + c0);
    v8h hi = *(const v8h*)(rowPtr + c0 + 8);
    v16h a;
#pragma unroll
    for (int i = 0; i < 8; ++i) { a[i] = lo[i]; a[i + 8] = hi[i]; }
    return a;
}

// ---------------- fused per-molecule GNN ----------------
__global__ __launch_bounds__(256) void gnn_mol_kernel(
    const float* __restrict__ adj, const int* __restrict__ fp,
    const float* __restrict__ embed,
    const _Float16* __restrict__ Wf16, const float* __restrict__ Wf_b,
    _Float16* __restrict__ x16, int outCol)
{
    __shared__ __align__(32) _Float16 sV[ATOMSV][DIMV];   // layer input (f16)
    __shared__ __align__(32) _Float16 sH[ATOMSV][DIMV];   // relu(VW^T+b) (f16)
    __shared__ __align__(32) _Float16 sA[ATOMSV][ATOMSV]; // adjacency block
    __shared__ __align__(16) float sU[ATOMSV][DIMV];      // h + adj*h (f32);
                                                          // also TDM staging
#if HAVE_ASYNC
    __shared__ __align__(16) float sStage[ATOMSV][DIMV];  // async gather stage
#endif

    const int m    = blockIdx.x;
    const int tid  = threadIdx.x;
    const int lane = tid & 31;
    const int wave = tid >> 5;

#if HAVE_TDM
    // Tensor Data Mover: DMA the 32x32 f32 block-diagonal adjacency tile into
    // LDS (staged in sU). One descriptor, issued by wave 0 only (EXEC ignored).
    if (wave == 0) {
        unsigned long long gaddr =
            (unsigned long long)(uintptr_t)(adj + (size_t)(m * ATOMSV) * N_ATV
                                                + m * ATOMSV);
        unsigned int laddr = (unsigned int)(uintptr_t)(&sU[0][0]);
        v4u g0;
        g0[0] = 1u;                                   // count=1 (valid), user
        g0[1] = laddr;                                // lds_addr (bytes)
        g0[2] = (unsigned int)gaddr;                  // global_addr[31:0]
        g0[3] = (unsigned int)((gaddr >> 32) & 0x1FFFFFFu) | (2u << 30); // type=2
        v8i g1;
        g1[0] = (int)(2u << 16);        // workgroup_mask=0, data_size=2 (4B)
        g1[1] = (int)(4096u << 16);     // tensor_dim0[15:0]=4096 in [31:16]
        g1[2] = (int)(4096u << 16);     // dim0 hi=0; tensor_dim1[15:0]=4096
        g1[3] = (int)(32u << 16);       // dim1 hi=0; tile_dim0=32
        g1[4] = 32;                     // tile_dim1=32; tile_dim2=0
        g1[5] = 4096;                   // tensor_dim0_stride[31:0]=4096 elems
        g1[6] = 0;                      // stride hi / dim1_stride lo
        g1[7] = 0;
        v4i gz4 = {0, 0, 0, 0};         // groups 2/3 unused (2D tensor)
        v8i gz8 = {0, 0, 0, 0, 0, 0, 0, 0};
        __builtin_amdgcn_tensor_load_to_lds(g0, g1, gz4, gz4, gz8, 0);
    }
#endif

#if HAVE_ASYNC
    // Async global->LDS gather of this molecule's embeddings (32x64 f32).
    for (int e = tid; e < ATOMSV * DIMV; e += 256) {
        int r = e >> 6, c = e & 63;
        const float* gp = embed + (size_t)fp[m * ATOMSV + r] * DIMV + c;
        __builtin_amdgcn_global_load_async_to_lds_b32(
            (__attribute__((address_space(1))) int*)gp,
            (__attribute__((address_space(3))) int*)&sStage[r][c], 0, 0);
    }
#if __has_builtin(__builtin_amdgcn_s_wait_asynccnt)
    __builtin_amdgcn_s_wait_asynccnt(0);
#else
    asm volatile("s_wait_asynccnt 0x0" ::: "memory");
#endif
#else
    for (int e = tid; e < ATOMSV * DIMV; e += 256) {
        int r = e >> 6, c = e & 63;
        sV[r][c] = (_Float16)embed[(size_t)fp[m * ATOMSV + r] * DIMV + c];
    }
#endif

#if HAVE_TDM
    if (wave == 0) __builtin_amdgcn_s_wait_tensorcnt((short)0);
    __syncthreads();
    {   // convert staged f32 adjacency tile -> f16 sA
        const float* stage = &sU[0][0];
        for (int e = tid; e < ATOMSV * ATOMSV; e += 256)
            sA[e >> 5][e & 31] = (_Float16)stage[e];
    }
#else
    for (int e = tid; e < ATOMSV * ATOMSV; e += 256) {
        int i = e >> 5, j = e & 31;
        sA[i][j] = (_Float16)adj[(size_t)(m * ATOMSV + i) * N_ATV + m * ATOMSV + j];
    }
#endif
#if HAVE_ASYNC
    __syncthreads();
    for (int e = tid; e < ATOMSV * DIMV; e += 256)
        sV[e >> 6][e & 63] = (_Float16)sStage[e >> 6][e & 63];
#endif
    __syncthreads();

    const int mT  = wave >> 2;        // 2 M-tiles of 16 rows
    const int nT  = wave & 3;         // 4 N-tiles of 16 cols
    const int m0  = mT * 16, n0 = nT * 16;
    const int hi  = lane >> 4;        // lane half (K-chunk select)
    const int l15 = lane & 15;
    const int n   = n0 + l15;         // output column owned by this lane

    for (int l = 0; l < NLAYER; ++l) {
        const _Float16* Wl = Wf16 + l * DIMV * DIMV;
        const float*    bl = Wf_b + l * DIMV;

        // H = relu(V @ W^T + b): K=64 -> two 16x16x32 WMMAs.
        v8f acc = {};
#pragma unroll
        for (int ks = 0; ks < 2; ++ks) {
            v16h a = load_a_frag(&sV[m0 + l15][0], ks * 32 + hi * 16);
            // B[k][j] = W[n0+j][k] -> contiguous 16 halfs of weight row.
            v16h b = *(const v16h*)(Wl + (size_t)(n0 + l15) * DIMV + ks * 32 + hi * 16);
            acc = __builtin_amdgcn_wmma_f32_16x16x32_f16(
                false, a, false, b, (short)0, acc, false, false);
        }
        const float bias = bl[n];
#pragma unroll
        for (int r = 0; r < 8; ++r) {
            int row = m0 + r + 8 * hi;
            float h = fmaxf(acc[r] + bias, 0.0f);
            sH[row][n] = (_Float16)h;
            sU[row][n] = h;
        }
        __syncthreads();

        // U += adj @ H : one 16x16x32 WMMA per tile (K = 32 atoms).
        v16h aA = load_a_frag(&sA[m0 + l15][0], hi * 16);
        v16h bH;
#pragma unroll
        for (int e = 0; e < 16; ++e) bH[e] = sH[e + 16 * hi][n];
        v8f acc2 = {};
        acc2 = __builtin_amdgcn_wmma_f32_16x16x32_f16(
            false, aA, false, bH, (short)0, acc2, false, false);
#pragma unroll
        for (int r = 0; r < 8; ++r) sU[m0 + r + 8 * hi][n] += acc2[r];
        __syncthreads();

        // Row-wise l2 normalize: each wave owns 4 rows, wave32 shfl reduce.
        for (int r = wave * 4; r < wave * 4 + 4; ++r) {
            float x0 = sU[r][lane];
            float x1 = sU[r][lane + 32];
            float s = x0 * x0 + x1 * x1;
#pragma unroll
            for (int off = 16; off > 0; off >>= 1) s += __shfl_xor(s, off, 32);
            float inv = 1.0f / fmaxf(sqrtf(s), 1e-12f);
            x0 *= inv; x1 *= inv;
            sV[r][lane] = (_Float16)x0; sV[r][lane + 32] = (_Float16)x1;
            sU[r][lane] = x0;           sU[r][lane + 32] = x1;
        }
        __syncthreads();
    }

    // Sum-pool 32 atoms -> molecule feature row of concatenated x.
    if (tid < DIMV) {
        float s = 0.0f;
#pragma unroll
        for (int i = 0; i < ATOMSV; ++i) s += sU[i][tid];
        x16[(size_t)m * FPADV + outCol + tid] = (_Float16)s;
    }
}

// ---------------- output MLP layer (187 padded to 192) ----------------
__global__ __launch_bounds__(256) void mlp_layer_kernel(
    const _Float16* __restrict__ xin, const _Float16* __restrict__ W16,
    const float* __restrict__ bias, _Float16* __restrict__ xout)
{
    const int tid  = threadIdx.x;
    const int lane = tid & 31;
    const int wave = tid >> 5;
    const int g    = blockIdx.x * 8 + wave;   // 96 tiles: 8 M x 12 N
    const int mT   = g & 7;
    const int nT   = g >> 3;
    const int m0   = mT * 16, n0 = nT * 16;
    const int hi   = lane >> 4;
    const int l15  = lane & 15;

    v8f acc = {};
#pragma unroll
    for (int ks = 0; ks < 6; ++ks) {          // K = 192 (zeros past 187)
        v16h a = load_a_frag(xin + (size_t)(m0 + l15) * FPADV, ks * 32 + hi * 16);
        v16h b = *(const v16h*)(W16 + (size_t)(n0 + l15) * FPADV + ks * 32 + hi * 16);
        acc = __builtin_amdgcn_wmma_f32_16x16x32_f16(
            false, a, false, b, (short)0, acc, false, false);
    }
    const int n = n0 + l15;
    const float bv = (n < FEATV) ? bias[n] : 0.0f;
#pragma unroll
    for (int r = 0; r < 8; ++r) {
        int row = m0 + r + 8 * hi;
        float h = fmaxf(acc[r] + bv, 0.0f);
        xout[(size_t)row * FPADV + n] = (n < FEATV) ? (_Float16)h : (_Float16)0.0f;
    }
}

// ---------------- prediction + MSE loss ----------------
__global__ __launch_bounds__(128) void pred_loss_kernel(
    const _Float16* __restrict__ x2, const float* __restrict__ Wp_w,
    const float* __restrict__ Wp_b, const float* __restrict__ cv,
    float* __restrict__ out)
{
    __shared__ float red[128];
    const int m = threadIdx.x;
    float acc = Wp_b[0];
    for (int c = 0; c < FEATV; ++c)
        acc += (float)x2[(size_t)m * FPADV + c] * Wp_w[c];
    float d = acc - cv[m];
    red[m] = d * d;
    __syncthreads();
    for (int s = 64; s > 0; s >>= 1) {
        if (m < s) red[m] += red[m + s];
        __syncthreads();
    }
    if (m == 0) out[0] = red[0] / (float)N_MOLV;
}

// ---------------- weight / input staging ----------------
__global__ void cvt_f32_to_f16_kernel(const float* __restrict__ src,
                                      _Float16* __restrict__ dst, int nElem) {
    int i = blockIdx.x * blockDim.x + threadIdx.x;
    if (i < nElem) dst[i] = (_Float16)src[i];
}

__global__ void cvt_wo_pad_kernel(const float* __restrict__ Wo,
                                  _Float16* __restrict__ W16) {
    int idx = blockIdx.x * blockDim.x + threadIdx.x;
    if (idx >= 2 * FPADV * FPADV) return;
    int l = idx / (FPADV * FPADV);
    int rem = idx % (FPADV * FPADV);
    int r = rem / FPADV, c = rem % FPADV;
    float v = (r < FEATV && c < FEATV)
                  ? Wo[(size_t)l * FEATV * FEATV + (size_t)r * FEATV + c]
                  : 0.0f;
    W16[idx] = (_Float16)v;
}

__global__ void fill_x_scalars_kernel(
    const float* __restrict__ homoD, const float* __restrict__ lumoD,
    const float* __restrict__ homoA, const float* __restrict__ lumoA,
    const float* __restrict__ desc, _Float16* __restrict__ x16)
{
    int idx = blockIdx.x * blockDim.x + threadIdx.x;
    if (idx >= N_MOLV * FPADV) return;
    int m = idx / FPADV, c = idx % FPADV;
    if (c < DIMV) return;                 // mol_D written by GNN kernel
    if (c >= 66 && c < 130) return;       // mol_A written by GNN kernel
    float v;
    if (c == 64)        v = homoD[m];
    else if (c == 65)   v = lumoD[m];
    else if (c == 130)  v = homoA[m];
    else if (c == 131)  v = lumoA[m];
    else if (c >= 132 && c < FEATV) v = desc[m * 55 + (c - 132)];
    else v = 0.0f;                        // pad cols 187..191
    x16[idx] = (_Float16)v;
}

extern "C" void kernel_launch(void* const* d_in, const int* in_sizes, int n_in,
                              void* d_out, int out_size, void* d_ws, size_t ws_size,
                              hipStream_t stream) {
    (void)in_sizes; (void)n_in; (void)out_size; (void)ws_size;
    const float* adj_D  = (const float*)d_in[0];
    const float* adj_A  = (const float*)d_in[1];
    const int*   fp_D   = (const int*)d_in[2];
    const int*   fp_A   = (const int*)d_in[3];
    // d_in[4], d_in[5]: seg ids (implicit in block-diagonal layout; unused)
    const float* homo_D = (const float*)d_in[6];
    const float* lumo_D = (const float*)d_in[7];
    const float* homo_A = (const float*)d_in[8];
    const float* lumo_A = (const float*)d_in[9];
    const float* desc   = (const float*)d_in[10];
    const float* cv     = (const float*)d_in[11];
    const float* embed  = (const float*)d_in[12];
    const float* Wf_w   = (const float*)d_in[13];
    const float* Wf_b   = (const float*)d_in[14];
    const float* Wo_w   = (const float*)d_in[15];
    const float* Wo_b   = (const float*)d_in[16];
    const float* Wp_w   = (const float*)d_in[17];
    const float* Wp_b   = (const float*)d_in[18];

    // Workspace layout (all f16): x, two ping-pong activations, weights.
    _Float16* x16  = (_Float16*)d_ws;             // 128*192
    _Float16* y16a = x16  + N_MOLV * FPADV;       // 128*192
    _Float16* y16b = y16a + N_MOLV * FPADV;       // 128*192
    _Float16* Wf16 = y16b + N_MOLV * FPADV;       // 3*64*64
    _Float16* Wo16 = Wf16 + 3 * DIMV * DIMV;      // 2*192*192

    cvt_f32_to_f16_kernel<<<(3 * DIMV * DIMV + 255) / 256, 256, 0, stream>>>(
        Wf_w, Wf16, 3 * DIMV * DIMV);
    cvt_wo_pad_kernel<<<(2 * FPADV * FPADV + 255) / 256, 256, 0, stream>>>(
        Wo_w, Wo16);
    fill_x_scalars_kernel<<<(N_MOLV * FPADV + 255) / 256, 256, 0, stream>>>(
        homo_D, lumo_D, homo_A, lumo_A, desc, x16);

    gnn_mol_kernel<<<N_MOLV, 256, 0, stream>>>(adj_D, fp_D, embed, Wf16, Wf_b, x16, 0);
    gnn_mol_kernel<<<N_MOLV, 256, 0, stream>>>(adj_A, fp_A, embed, Wf16, Wf_b, x16, 66);

    mlp_layer_kernel<<<12, 256, 0, stream>>>(x16, Wo16, Wo_b, y16a);
    mlp_layer_kernel<<<12, 256, 0, stream>>>(y16a, Wo16 + FPADV * FPADV, Wo_b + FEATV, y16b);

    pred_loss_kernel<<<1, 128, 0, stream>>>(y16b, Wp_w, Wp_b, cv, (float*)d_out);
}
